// TfLSTM_7181185319293
// MI455X (gfx1250) — matrix operand order repaired and checked by reading, not compiled, not run on hardware
//
#include <hip/hip_runtime.h>
#include <hip/hip_bf16.h>
#include <math.h>

// ---------------------------------------------------------------------------
// Types for CDNA5 WMMA (wave32)
// ---------------------------------------------------------------------------
typedef __bf16 bf16;
typedef bf16  v8bf  __attribute__((ext_vector_type(8)));
typedef bf16  v16bf __attribute__((ext_vector_type(16)));
typedef float v8f   __attribute__((ext_vector_type(8)));

#define T_STEPS 512
#define HP      128      // padded hidden size for both layers
#define NCOLS   512      // 4 * HP
#define OUTDIM  19

__device__ __forceinline__ float sigm(float x) { return 1.0f / (1.0f + __expf(-x)); }

// Load 16 bf16 as two 16B chunks (p and p2 each 16B-aligned in LDS).
__device__ __forceinline__ v16bf load16(const bf16* p, const bf16* p2) {
    v8bf lo = *reinterpret_cast<const v8bf*>(p);
    v8bf hi = *reinterpret_cast<const v8bf*>(p2);
    v16bf r;
#pragma unroll
    for (int i = 0; i < 8; ++i) { r[i] = lo[i]; r[i + 8] = hi[i]; }
    return r;
}

// ---------------------------------------------------------------------------
// Repack kernels
// ---------------------------------------------------------------------------
// x f32 [256][512][64] -> bf16 (one-time, so layer-1 input prefetch is a raw copy)
__global__ void repack_x(const float* __restrict__ x, bf16* __restrict__ dst) {
    int idx = blockIdx.x * 256 + threadIdx.x;
    if (idx >= 256 * 512 * 64) return;
    dst[idx] = (bf16)x[idx];
}

// Layer 1 fused weights: K = 64 (x) + 128 (h pad) = 192 rows, 512 cols. Real H1 = 100.
__global__ void repack_wu1(const float* __restrict__ W1, const float* __restrict__ U1,
                           bf16* __restrict__ dst) {
    int idx = blockIdx.x * 256 + threadIdx.x;
    if (idx >= 192 * 512) return;
    int k = idx >> 9, col = idx & 511, G = col >> 7, j = col & 127;
    float v = 0.0f;
    if (k < 64) {
        if (j < 100) v = W1[k * 400 + G * 100 + j];
    } else {
        int kh = k - 64;
        if (kh < 100 && j < 100) v = U1[kh * 400 + G * 100 + j];
    }
    dst[idx] = (bf16)v;
}

// Layer 2 fused weights: K = 128 (h1 pad) + 128 (h2) = 256 rows, 512 cols. H2 = 128.
__global__ void repack_wu2(const float* __restrict__ W2, const float* __restrict__ U2,
                           bf16* __restrict__ dst) {
    int idx = blockIdx.x * 256 + threadIdx.x;
    if (idx >= 256 * 512) return;
    int k = idx >> 9, col = idx & 511;
    float v = 0.0f;
    if (k < 128) {
        if (k < 100) v = W2[k * 512 + col];
    } else {
        v = U2[(k - 128) * 512 + col];
    }
    dst[idx] = (bf16)v;
}

__global__ void repack_b1(const float* __restrict__ b1, float* __restrict__ dst) {
    int idx = blockIdx.x * 256 + threadIdx.x;
    if (idx >= 512) return;
    int G = idx >> 7, j = idx & 127;
    dst[idx] = (j < 100) ? b1[G * 100 + j] : 0.0f;
}

// ---------------------------------------------------------------------------
// Fused LSTM layer kernel: one workgroup = 16 batch rows, all 512 steps.
// Weights resident in LDS; input slice double-buffered + register-prefetched;
// h staging double-buffered so the whole step needs ONE barrier.
//   KIN  : padded input width (64 for L1, 128 for L2), stored width of in_bf
//   KTOT : KIN + 128
//   LAST : dense + softmax head at final step instead of writing h sequence
// ---------------------------------------------------------------------------
template <int KIN, int KTOT, bool LAST>
__global__ __launch_bounds__(256) void lstm_wmma(
    const bf16* __restrict__ in_bf, const bf16* __restrict__ wu,
    const float* __restrict__ bias, bf16* __restrict__ hseq_out,
    const float* __restrict__ Wd, const float* __restrict__ bd,
    float* __restrict__ out) {

    constexpr int KCH  = KTOT / 32;   // total K chunks
    constexpr int KCHI = KIN / 32;    // chunks sourced from the input buffer
    constexpr int NPF  = 16 * KIN / 8;   // prefetch uint4s per step (<= 256)

    __shared__ __align__(16) bf16  s_wu[KTOT * NCOLS];     // fused weights, resident
    __shared__ __align__(16) bf16  s_in[2][16 * KIN];      // input slice, double buffered
    __shared__ __align__(16) bf16  s_ah[2][16 * HP];       // h (bf16), double buffered
    __shared__ float s_c[16 * HP];
    __shared__ float s_h[16 * HP];                          // f32 h (final head)

    const int tid = threadIdx.x;
    const int b0  = blockIdx.x * 16;

    // ---- One-time setup ----
    for (int i = tid; i < KTOT * NCOLS / 8; i += 256)
        reinterpret_cast<v8bf*>(s_wu)[i] = reinterpret_cast<const v8bf*>(wu)[i];
    for (int i = tid; i < 16 * HP; i += 256) {
        s_c[i] = 0.0f; s_h[i] = 0.0f;
        s_ah[0][i] = (bf16)0.0f;
    }
    // Stage input slice for t = 0.
    const int mm_pf = tid / (KIN / 8);
    const int ko_pf = (tid % (KIN / 8)) * 8;
    if (tid < NPF) {
        const uint4* src = reinterpret_cast<const uint4*>(
            &in_bf[((size_t)(b0 + mm_pf) * T_STEPS + 0) * KIN + ko_pf]);
        *reinterpret_cast<uint4*>(&s_in[0][mm_pf * KIN + ko_pf]) = *src;
    }
    __syncthreads();

    const int w    = tid >> 5;       // wave id: owns hidden cols [16w, 16w+16)
    const int lane = tid & 31;
    const int m    = lane & 15;
    const int hf   = lane >> 4;      // lane half
    const int j    = w * 16 + m;     // this lane's hidden column

    // Loop-invariant per-lane bias (gate G, column j).
    float bv[4];
#pragma unroll
    for (int G = 0; G < 4; ++G) bv[G] = bias[G * 128 + j];

    int buf = 0;
    for (int t = 0; t < T_STEPS; ++t) {
        const int nbuf = buf ^ 1;

        // ---- Issue prefetch of next step's input (latency hidden by WMMAs) ----
        uint4 pf;
        const int tn = (t + 1) & (T_STEPS - 1);
        if (tid < NPF)
            pf = *reinterpret_cast<const uint4*>(
                &in_bf[((size_t)(b0 + mm_pf) * T_STEPS + tn) * KIN + ko_pf]);

        // ---- A fragments: input chunks from s_in[buf], h chunks from s_ah[buf] ----
        v16bf afr[KCH];
#pragma unroll
        for (int kc = 0; kc < KCH; ++kc) {
            const bf16* p = (kc < KCHI)
                ? &s_in[buf][m * KIN + kc * 32 + 8 * hf]
                : &s_ah[buf][m * HP + (kc - KCHI) * 32 + 8 * hf];
            afr[kc] = load16(p, p + 16);
        }

        // ---- z[G] = bias + A @ WU  for gates i,f,g,o of columns [16w,16w+16) ----
        v8f acc[4];
#pragma unroll
        for (int G = 0; G < 4; ++G) {
            v8f a;
#pragma unroll
            for (int r = 0; r < 8; ++r) a[r] = bv[G];
#pragma unroll
            for (int kc = 0; kc < KCH; ++kc) {
                // B fragment: lane = N column (m), 16 K values in table order (strided gather from LDS)
                v16bf bfr;
#pragma unroll
                for (int s = 0; s < 16; ++s)
                    bfr[s] = s_wu[(kc * 32 + (s < 8 ? 8 * hf + s : 16 + 8 * hf + (s - 8))) * NCOLS + (G * 8 + w) * 16 + m];
                a = __builtin_amdgcn_wmma_f32_16x16x32_bf16(
                        false, afr[kc], false, bfr, (short)0, a, false, false);
            }
            acc[G] = a;
        }

        // ---- Gates + state update; stage next-step h (bf16) directly ----
#pragma unroll
        for (int r = 0; r < 8; ++r) {
            int M = r + 8 * hf;
            float iv = sigm(acc[0][r]);
            float fv = sigm(acc[1][r]);
            float gv = fmaxf(acc[2][r], 0.0f);   // activation = relu
            float ov = sigm(acc[3][r]);
            float cn = fv * s_c[M * HP + j] + iv * gv;   // lane-private cells
            float hv = ov * fmaxf(cn, 0.0f);             // h = o * relu(c)
            s_c[M * HP + j] = cn;
            s_h[M * HP + j] = hv;
            s_ah[nbuf][M * HP + j] = (bf16)hv;
            if (!LAST)
                hseq_out[((size_t)(b0 + M) * T_STEPS + t) * HP + j] = (bf16)hv;
        }

        // ---- Land the prefetched input in the alternate buffer ----
        if (tid < NPF)
            *reinterpret_cast<uint4*>(&s_in[nbuf][mm_pf * KIN + ko_pf]) = pf;

        __syncthreads();   // single barrier: publishes s_ah[nbuf] + s_in[nbuf]
        buf = nbuf;
    }

    if (LAST) {
        __syncthreads();
        // Dense + softmax head on final h2 (tiny: 16 rows x 19 outputs).
        if (tid < 16) {
            int mm = tid;
            float lg[OUTDIM];
            float mx = -1e30f;
            for (int o = 0; o < OUTDIM; ++o) {
                float s = bd[o];
                for (int k = 0; k < HP; ++k) s += s_h[mm * HP + k] * Wd[k * OUTDIM + o];
                lg[o] = s;
                mx = fmaxf(mx, s);
            }
            float den = 0.0f;
            for (int o = 0; o < OUTDIM; ++o) { lg[o] = __expf(lg[o] - mx); den += lg[o]; }
            float inv = 1.0f / den;
            for (int o = 0; o < OUTDIM; ++o)
                out[(size_t)(b0 + mm) * OUTDIM + o] = lg[o] * inv;
        }
    }
}

// ---------------------------------------------------------------------------
// Launch
// ---------------------------------------------------------------------------
extern "C" void kernel_launch(void* const* d_in, const int* in_sizes, int n_in,
                              void* d_out, int out_size, void* d_ws, size_t ws_size,
                              hipStream_t stream) {
    const float* x  = (const float*)d_in[0];
    const float* W1 = (const float*)d_in[1];
    const float* U1 = (const float*)d_in[2];
    const float* b1 = (const float*)d_in[3];
    const float* W2 = (const float*)d_in[4];
    const float* U2 = (const float*)d_in[5];
    const float* b2 = (const float*)d_in[6];
    const float* Wd = (const float*)d_in[7];
    const float* bd = (const float*)d_in[8];
    float* out = (float*)d_out;

    // Workspace layout (all offsets 16B aligned):
    //   h1 bf16 [256][512][128] = 33,554,432 B
    //   x  bf16 [256][512][64]  = 16,777,216 B
    //   wu1 bf16 [192][512]     =    196,608 B
    //   wu2 bf16 [256][512]     =    262,144 B
    //   b1pad f32 [512]         =      2,048 B
    char* base = (char*)d_ws;
    bf16*  h1bf = (bf16*)base;
    bf16*  xbf  = (bf16*)(base + 33554432);
    bf16*  wu1  = (bf16*)(base + 50331648);
    bf16*  wu2  = (bf16*)(base + 50331648 + 196608);
    float* bp1  = (float*)(base + 50331648 + 196608 + 262144);

    repack_x  <<<(256 * 512 * 64 + 255) / 256, 256, 0, stream>>>(x, xbf);
    repack_wu1<<<(192 * 512 + 255) / 256, 256, 0, stream>>>(W1, U1, wu1);
    repack_wu2<<<(256 * 512 + 255) / 256, 256, 0, stream>>>(W2, U2, wu2);
    repack_b1 <<<2, 256, 0, stream>>>(b1, bp1);

    // Layer 1: input x (bf16, KIN=64), writes padded-bf16 hidden sequence.
    lstm_wmma<64, 192, false><<<16, 256, 0, stream>>>(
        xbf, wu1, bp1, h1bf, nullptr, nullptr, nullptr);

    // Layer 2: input h1 (bf16, KIN=128), ends with dense + softmax head.
    lstm_wmma<128, 256, true><<<16, 256, 0, stream>>>(
        h1bf, wu2, b2, nullptr, Wd, bd, out);
}
